// QuantMultiHeadSelfAttention_81922206204432
// MI455X (gfx1250) — 16-bit and 8-bit matrix operand order repaired and checked by reading, compiled, not run on hardware
//
#include <hip/hip_runtime.h>
#include <math.h>
#include <stdint.h>


typedef __attribute__((ext_vector_type(16))) _Float16 v16h;
typedef __attribute__((ext_vector_type(8)))  _Float16 v8h;
typedef __attribute__((ext_vector_type(4)))  _Float16 v4h;
typedef __attribute__((ext_vector_type(8)))  float    v8f;
typedef __attribute__((ext_vector_type(8)))  int      v8i;

#define NHEADS 12
#define DIM    768
#define DPH    64
#define SEQ    1024
#define BS     8
#define MROWS  (BS * SEQ)
#define BH     (BS * NHEADS)
#define NEGINF (-3.402823466e+38f)

__device__ __forceinline__ v8f wmma_f16(v16h a, v16h b, v8f c) {
  return __builtin_amdgcn_wmma_f32_16x16x32_f16(false, a, false, b, (short)0, c,
                                                false, false);
}
template <bool SA, bool SB>
__device__ __forceinline__ v8i wmma_iu8(v8i a, v8i b, v8i c) {
  return __builtin_amdgcn_wmma_i32_16x16x64_iu8(SA, a, SB, b, c, false, false);
}

__device__ __forceinline__ float wave_max32(float v) {
#pragma unroll
  for (int m = 16; m >= 1; m >>= 1) v = fmaxf(v, __shfl_xor(v, m, 32));
  return v;
}
__device__ __forceinline__ void atomic_max_abs(float* addr, float v) {
  atomicMax((unsigned int*)addr, __float_as_uint(v));
}
__device__ __forceinline__ float qscale(float absmax) {
  return fmaxf(absmax * (1.0f / 127.0f), 1e-12f);
}

__device__ __forceinline__ v16h load_a16(const _Float16* __restrict__ src,
                                         int ld, int row0, int k0, int lane) {
  const int half = lane >> 4, r = lane & 15;
  const _Float16* p = src + (size_t)(row0 + r) * ld + k0 + 8 * half;
  const v8h lo = *(const v8h*)p;
  const v8h hi = *(const v8h*)(p + 16);
  v16h f;
#pragma unroll
  for (int i = 0; i < 8; ++i) { f[i] = lo[i]; f[8 + i] = hi[i]; }
  return f;
}
__device__ __forceinline__ v16h load_b16(const _Float16* __restrict__ src,
                                         int ld, int col0, int k0, int lane) {
  const int kh = lane >> 4, n = lane & 15;
  const _Float16* p = src + (size_t)(col0 + n) * ld + k0 + 8 * kh;
  const v8h lo = *(const v8h*)p;
  const v8h hi = *(const v8h*)(p + 16);
  v16h f;
#pragma unroll
  for (int i = 0; i < 8; ++i) { f[i] = lo[i]; f[8 + i] = hi[i]; }
  return f;
}
__device__ __forceinline__ v8i load_a8(const char* __restrict__ src, int ld,
                                       int row0, int k0, int lane) {
  const int half = lane >> 4, r = lane & 15;
  const char* p = src + (size_t)(row0 + r) * ld + k0 + 8 * half;
  v8i f;
#pragma unroll
  for (int c = 0; c < 4; ++c) {
    const int2 d = *(const int2*)(p + c * 16);
    f[2 * c] = d.x;
    f[2 * c + 1] = d.y;
  }
  return f;
}
__device__ __forceinline__ v8i load_b8(const char* __restrict__ src, int ld,
                                       int col0, int k0, int lane) {
  const int kh = lane >> 4, n = lane & 15;
  const char* p = src + (size_t)(col0 + n) * ld + k0 + 8 * kh;
  v8i f;
#pragma unroll
  for (int c = 0; c < 4; ++c) {
    const int2 d = *(const int2*)(p + c * 16);
    f[2 * c] = d.x;
    f[2 * c + 1] = d.y;
  }
  return f;
}

__global__ void zero_scalars_kernel(float* __restrict__ s) {
  if (threadIdx.x < 8) s[threadIdx.x] = 0.0f;
}

__global__ void cvt_f16_kernel(const float4* __restrict__ src,
                               v4h* __restrict__ dst, int n4) {
  int i = blockIdx.x * blockDim.x + threadIdx.x;
  const int stride = gridDim.x * blockDim.x;
  for (; i < n4; i += stride) {
    const float4 v = src[i];
    dst[i] = (v4h){(_Float16)v.x, (_Float16)v.y, (_Float16)v.z, (_Float16)v.w};
  }
}

__global__ void quant_f16_kernel(const float4* __restrict__ src,
                                 v4h* __restrict__ dst, int n4,
                                 const float* __restrict__ absmax) {
  const float sc = qscale(absmax[0]);
  const float inv = 1.0f / sc;
  int i = blockIdx.x * blockDim.x + threadIdx.x;
  const int stride = gridDim.x * blockDim.x;
  for (; i < n4; i += stride) {
    const float4 v = src[i];
    dst[i] = (v4h){(_Float16)(rintf(v.x * inv) * sc),
                   (_Float16)(rintf(v.y * inv) * sc),
                   (_Float16)(rintf(v.z * inv) * sc),
                   (_Float16)(rintf(v.w * inv) * sc)};
  }
}

__global__ void quant_i8_kernel(const float4* __restrict__ src,
                                char4* __restrict__ dst, int n4,
                                const float* __restrict__ absmax) {
  const float inv = 1.0f / qscale(absmax[0]);
  int i = blockIdx.x * blockDim.x + threadIdx.x;
  const int stride = gridDim.x * blockDim.x;
  for (; i < n4; i += stride) {
    const float4 v = src[i];
    char4 o;
    o.x = (char)(int)rintf(v.x * inv);
    o.y = (char)(int)rintf(v.y * inv);
    o.z = (char)(int)rintf(v.z * inv);
    o.w = (char)(int)rintf(v.w * inv);
    dst[i] = o;
  }
}

__global__ void quant_vT_i8_kernel(const float* __restrict__ src,
                                   char4* __restrict__ dst,
                                   const float* __restrict__ absmax) {
  const float inv = 1.0f / qscale(absmax[0]);
  const int n4 = (int)((size_t)MROWS * DIM / 4);
  int i = blockIdx.x * blockDim.x + threadIdx.x;
  const int stride = gridDim.x * blockDim.x;
  for (; i < n4; i += stride) {
    const int s4   = i & (SEQ / 4 - 1);
    const int rest = i >> 8;
    const int d    = rest & (DPH - 1);
    const int bh   = rest >> 6;
    const int b = bh / NHEADS, h = bh % NHEADS;
    const float* p = src + ((size_t)b * SEQ + s4 * 4) * DIM + h * DPH + d;
    char4 o;
    o.x = (char)(int)rintf(p[0] * inv);
    o.y = (char)(int)rintf(p[(size_t)DIM] * inv);
    o.z = (char)(int)rintf(p[(size_t)2 * DIM] * inv);
    o.w = (char)(int)rintf(p[(size_t)3 * DIM] * inv);
    dst[i] = o;
  }
}

__global__ __launch_bounds__(128) void gemm_f16_kernel(
    const _Float16* __restrict__ A16, const _Float16* __restrict__ W16,
    const float* __restrict__ bias, float* __restrict__ C, float scale,
    float* absmax) {
  __shared__ __align__(16) _Float16 Bs[64 * 32];
  const int lane = threadIdx.x & 31;
  const int wave = threadIdx.x >> 5;
  const int half = lane >> 4, r = lane & 15;
  const int m0 = blockIdx.y * 64 + wave * 16;
  const int n0 = blockIdx.x * 64;

  v8f acc[4] = {};
  for (int k0 = 0; k0 < DIM; k0 += 32) {
    __syncthreads();
#pragma unroll
    for (int u = threadIdx.x; u < 256; u += 128) {
      const int n = u >> 2, c = (u & 3) * 8;
      const unsigned lofs = (unsigned)(size_t)&Bs[n * 32 + c];
      const _Float16* g = &W16[(size_t)(n0 + n) * DIM + k0 + c];
      asm volatile("global_load_async_to_lds_b128 %0, %1, off"
                   :
                   : "v"(lofs), "v"(g)
                   : "memory");
    }
    asm volatile("s_wait_asynccnt 0x0" ::: "memory");
    __syncthreads();
    if (k0 + 32 < DIM)
      __builtin_prefetch(&A16[(size_t)(m0 + r) * DIM + k0 + 32], 0, 3);
    const v16h a = load_a16(A16, DIM, m0, k0, lane);
#pragma unroll
    for (int nt = 0; nt < 4; ++nt)
      acc[nt] = wmma_f16(a, load_b16(Bs, 32, nt * 16, 0, lane), acc[nt]);
  }

  float amax = 0.0f;
#pragma unroll
  for (int nt = 0; nt < 4; ++nt) {
    const float bn = bias[n0 + nt * 16 + r];
#pragma unroll
    for (int j = 0; j < 8; ++j) {
      const float v = (acc[nt][j] + bn) * scale;
      amax = fmaxf(amax, fabsf(v));
      C[(size_t)(m0 + j + 8 * half) * DIM + n0 + nt * 16 + r] = v;
    }
  }
  if (absmax) {
    amax = wave_max32(amax);
    if (lane == 0) atomic_max_abs(absmax, amax);
  }
}

__global__ __launch_bounds__(128) void score_absmax_kernel(
    const char* __restrict__ Qi, const char* __restrict__ Ki,
    const float* scalars, float* absmaxS) {
  const int lane = threadIdx.x & 31;
  const int wave = threadIdx.x >> 5;
  const int bh = blockIdx.y;
  const int b = bh / NHEADS, h = bh % NHEADS;
  const int t = blockIdx.x * 4 + wave;
  const int qt = t >> 4;
  const int kt0 = (t & 15) * 4;
  const char* Qh = Qi + (size_t)b * SEQ * DIM + h * DPH;
  const char* Kh = Ki + (size_t)b * SEQ * DIM + h * DPH;
  const float sQK = qscale(scalars[0]) * qscale(scalars[1]);

  const v8i a = load_a8(Qh, DIM, qt * 16, 0, lane);
  int imax = 0;
#pragma unroll
  for (int kk = 0; kk < 4; ++kk) {
    v8i c = {};
    c = wmma_iu8<true, true>(a, load_b8(Kh, DIM, (kt0 + kk) * 16, 0, lane), c);
#pragma unroll
    for (int j = 0; j < 8; ++j) {
      int v = c[j];
      imax = max(imax, v < 0 ? -v : v);
    }
  }
  float amax = wave_max32((float)imax * sQK);
  if (lane == 0) atomic_max_abs(absmaxS, amax);
}

__global__ __launch_bounds__(128) void softmax_stats_kernel(
    const char* __restrict__ Qi, const char* __restrict__ Ki,
    const int* __restrict__ mask, const float* scalars,
    float* __restrict__ rowmax, float* __restrict__ denom, float* wmax) {
  __shared__ __align__(8) char strip[16 * SEQ];
  const int lane = threadIdx.x & 31;
  const int wave = threadIdx.x >> 5;
  const int half = lane >> 4, r = lane & 15;
  const int bh = blockIdx.y, qt = blockIdx.x;
  const int b = bh / NHEADS, h = bh % NHEADS;
  const char* Qh = Qi + (size_t)b * SEQ * DIM + h * DPH;
  const char* Kh = Ki + (size_t)b * SEQ * DIM + h * DPH;
  const float sQK = qscale(scalars[0]) * qscale(scalars[1]);
  const float sS = qscale(scalars[3]);
  const float invS = 1.0f / sS;

  const v8i a = load_a8(Qh, DIM, qt * 16, 0, lane);
  for (int kt = wave * 16; kt < wave * 16 + 16; ++kt) {
    v8i c = {};
    c = wmma_iu8<true, true>(a, load_b8(Kh, DIM, kt * 16, 0, lane), c);
    const int col = kt * 16 + r;
#pragma unroll
    for (int j = 0; j < 8; ++j)
      strip[(j + 8 * half) * SEQ + col] =
          (char)(int)rintf((float)c[j] * sQK * invS);
  }
  __syncthreads();

  const int row = threadIdx.x >> 3;
  const int sub = threadIdx.x & 7;
  const int* mrow = mask + b * SEQ;
  float rm = NEGINF;
  for (int c = sub; c < SEQ; c += 8)
    if (mrow[c] != 0) rm = fmaxf(rm, (float)strip[row * SEQ + c] * sS);
#pragma unroll
  for (int m = 4; m >= 1; m >>= 1) rm = fmaxf(rm, __shfl_xor(rm, m, 32));
  float sum = 0.0f;
  for (int c = sub; c < SEQ; c += 8)
    if (mrow[c] != 0) sum += __expf((float)strip[row * SEQ + c] * sS - rm);
#pragma unroll
  for (int m = 4; m >= 1; m >>= 1) sum += __shfl_xor(sum, m, 32);
  if (sub == 0) {
    const size_t idx = (size_t)bh * SEQ + qt * 16 + row;
    const float d = (sum > 0.0f) ? sum : 1.0f;
    rowmax[idx] = (rm > NEGINF) ? rm : 0.0f;
    denom[idx] = d;
    atomic_max_abs(wmax, 1.0f / d);
  }
}

__global__ __launch_bounds__(128) void context_kernel(
    const char* __restrict__ Qi, const char* __restrict__ Ki,
    const char* __restrict__ Vt, const int* __restrict__ mask,
    const float* scalars, const float* __restrict__ rowmax,
    const float* __restrict__ denom, float* __restrict__ Ctx,
    float* absmaxCtx) {
  __shared__ __align__(16) unsigned char wstrip[16 * SEQ];
  const int lane = threadIdx.x & 31;
  const int wave = threadIdx.x >> 5;
  const int half = lane >> 4, r = lane & 15;
  const int bh = blockIdx.y, qt = blockIdx.x;
  const int b = bh / NHEADS, h = bh % NHEADS;
  const char* Qh = Qi + (size_t)b * SEQ * DIM + h * DPH;
  const char* Kh = Ki + (size_t)b * SEQ * DIM + h * DPH;
  const char* Vh = Vt + (size_t)bh * DPH * SEQ;
  const float sQK = qscale(scalars[0]) * qscale(scalars[1]);
  const float sS = qscale(scalars[3]);
  const float invS = 1.0f / sS;
  const float wsc = qscale(scalars[4]);
  const float invW = 1.0f / wsc;
  const float sV = qscale(scalars[2]);
  const int* mrow = mask + b * SEQ;

  float rmv[8], dnv[8];
#pragma unroll
  for (int j = 0; j < 8; ++j) {
    const size_t idx = (size_t)bh * SEQ + qt * 16 + j + 8 * half;
    rmv[j] = rowmax[idx];
    dnv[j] = denom[idx];
  }

  const v8i a = load_a8(Qh, DIM, qt * 16, 0, lane);
  for (int kt = wave * 16; kt < wave * 16 + 16; ++kt) {
    v8i c = {};
    c = wmma_iu8<true, true>(a, load_b8(Kh, DIM, kt * 16, 0, lane), c);
    const int col = kt * 16 + r;
    const bool valid = (mrow[col] != 0);
#pragma unroll
    for (int j = 0; j < 8; ++j) {
      unsigned char lw = 0;
      if (valid) {
        const float qs = rintf((float)c[j] * sQK * invS) * sS;
        const float w = __expf(qs - rmv[j]) / dnv[j];
        lw = (unsigned char)(int)rintf(w * invW);
      }
      wstrip[(j + 8 * half) * SEQ + col] = lw;
    }
  }
  __syncthreads();

  v8i acc = {};
  const int n0 = wave * 16;
  for (int k0 = 0; k0 < SEQ; k0 += 64) {
    const v8i aw = load_a8((const char*)wstrip, SEQ, 0, k0, lane);
    const v8i bv = load_b8(Vh, SEQ, n0, k0, lane);
    acc = wmma_iu8<false, true>(aw, bv, acc);
  }
  const float cs = wsc * sV;
  float amax = 0.0f;
  const size_t row0 = (size_t)b * SEQ + qt * 16;
#pragma unroll
  for (int j = 0; j < 8; ++j) {
    const float v = (float)acc[j] * cs;
    amax = fmaxf(amax, fabsf(v));
    Ctx[(row0 + j + 8 * half) * DIM + h * DPH + n0 + r] = v;
  }
  amax = wave_max32(amax);
  if (lane == 0) atomic_max_abs(absmaxCtx, amax);
}

extern "C" void kernel_launch(void* const* d_in, const int* in_sizes, int n_in,
                              void* d_out, int out_size, void* d_ws,
                              size_t ws_size, hipStream_t stream) {
  (void)in_sizes; (void)n_in; (void)out_size; (void)ws_size;
  const float* query = (const float*)d_in[0];
  const float* key   = (const float*)d_in[1];
  const float* value = (const float*)d_in[2];
  const int*   mask  = (const int*)d_in[3];
  const float* Wq = (const float*)d_in[4];
  const float* bq = (const float*)d_in[5];
  const float* Wk = (const float*)d_in[6];
  const float* bk = (const float*)d_in[7];
  const float* Wv = (const float*)d_in[8];
  const float* bv = (const float*)d_in[9];
  const float* Wo = (const float*)d_in[10];
  const float* bo = (const float*)d_in[11];
  float* out = (float*)d_out;

  char* base = (char*)d_ws;
  const size_t NP = (size_t)MROWS * DIM;
  float*    F0  = (float*)(base);
  float*    F1  = (float*)(base + NP * 4);
  float*    F2  = (float*)(base + NP * 8);
  char*     I0  = (char*)(base + NP * 12);
  char*     I1  = (char*)(base + NP * 13);
  char*     I2  = (char*)(base + NP * 14);
  _Float16* H3  = (_Float16*)(base + NP * 15);
  _Float16* H4  = (_Float16*)(base + NP * 17);
  char*     tail = base + NP * 17 + (size_t)DIM * DIM * 2;
  float*    rowmax = (float*)(tail);
  float*    denom  = (float*)(tail + (size_t)BH * SEQ * 4);
  float*    scal   = (float*)(tail + (size_t)BH * SEQ * 8);

  zero_scalars_kernel<<<1, 32, 0, stream>>>(scal);

  const int n4x = (int)(NP / 4);
  const int n4w = (int)((size_t)DIM * DIM / 4);
  const dim3 ggemm(DIM / 64, MROWS / 64);

  cvt_f16_kernel<<<2048, 256, 0, stream>>>((const float4*)query, (v4h*)H3, n4x);
  cvt_f16_kernel<<<512, 256, 0, stream>>>((const float4*)Wq, (v4h*)H4, n4w);
  gemm_f16_kernel<<<ggemm, 128, 0, stream>>>(H3, H4, bq, F0, 0.125f, scal + 0);
  cvt_f16_kernel<<<2048, 256, 0, stream>>>((const float4*)key, (v4h*)H3, n4x);
  cvt_f16_kernel<<<512, 256, 0, stream>>>((const float4*)Wk, (v4h*)H4, n4w);
  gemm_f16_kernel<<<ggemm, 128, 0, stream>>>(H3, H4, bk, F1, 1.0f, scal + 1);
  cvt_f16_kernel<<<2048, 256, 0, stream>>>((const float4*)value, (v4h*)H3, n4x);
  cvt_f16_kernel<<<512, 256, 0, stream>>>((const float4*)Wv, (v4h*)H4, n4w);
  gemm_f16_kernel<<<ggemm, 128, 0, stream>>>(H3, H4, bv, F2, 1.0f, scal + 2);

  quant_i8_kernel<<<2048, 256, 0, stream>>>((const float4*)F0, (char4*)I0, n4x, scal + 0);
  quant_i8_kernel<<<2048, 256, 0, stream>>>((const float4*)F1, (char4*)I1, n4x, scal + 1);
  quant_vT_i8_kernel<<<2048, 256, 0, stream>>>(F2, (char4*)I2, scal + 2);

  score_absmax_kernel<<<dim3(256, BH), 128, 0, stream>>>(I0, I1, scal, scal + 3);
  softmax_stats_kernel<<<dim3(SEQ / 16, BH), 128, 0, stream>>>(
      I0, I1, mask, scal, rowmax, denom, scal + 4);
  context_kernel<<<dim3(SEQ / 16, BH), 128, 0, stream>>>(
      I0, I1, I2, mask, scal, rowmax, denom, F0, scal + 5);

  quant_f16_kernel<<<2048, 256, 0, stream>>>((const float4*)F0, (v4h*)H3, n4x, scal + 5);
  cvt_f16_kernel<<<512, 256, 0, stream>>>((const float4*)Wo, (v4h*)H4, n4w);
  gemm_f16_kernel<<<ggemm, 128, 0, stream>>>(H3, H4, bo, out, 1.0f, nullptr);
}
